// MambaFPNFusion_10857677324812
// MI455X (gfx1250) — matrix operand order repaired and checked by reading, not compiled, not run on hardware
//
#include <hip/hip_runtime.h>
#include <hip/hip_bf16.h>

// ---------------------------------------------------------------------------
// Types for CDNA5 WMMA (wave32, 16x16x32 f16 -> f32 accumulate)
// ---------------------------------------------------------------------------
typedef __attribute__((ext_vector_type(16))) _Float16 v16h;
typedef __attribute__((ext_vector_type(8)))  _Float16 v8h;
typedef __attribute__((ext_vector_type(2)))  _Float16 v2h;
typedef __attribute__((ext_vector_type(8)))  float    v8f;

union Frag {
    v16h f;
    v8h  h[2];
};

// ---------------------------------------------------------------------------
// Generic strided/batched GEMM: C[m,n] = sum_k A[m,k]*B[k,n]
// A/B read as f32 via (row-stride, col-stride), converted to f16 in LDS.
// Block: 128 threads (4 waves), 64x64 C tile; each wave a 32x32 sub-tile via
// 2x2 v_wmma_f32_16x16x32_f16. Branchless clamped loads on edge tiles only.
// ---------------------------------------------------------------------------
#define GT_M 64
#define GT_N 64
#define GT_K 32
#define LDH  40   // halves per LDS row (80B: 16B-aligned, skewed vs 64 banks)

__global__ __launch_bounds__(128)
void gemm_wmma_kernel(const float* __restrict__ A, const float* __restrict__ B,
                      float* __restrict__ C, int M, int N, int K,
                      int rsA, int csA, int rsB, int csB,
                      int rsC, int csC,
                      long bsA, long bsB, long bsC)
{
    __shared__ __align__(16) _Float16 sA[GT_M * LDH];
    __shared__ __align__(16) _Float16 sB[GT_N * LDH];

    const int tid  = threadIdx.x;
    const int lane = tid & 31;
    const int wave = tid >> 5;
    const int wm   = (wave >> 1) & 1;   // wave row (0..1) -> 32 rows
    const int wn   = wave & 1;          // wave col (0..1) -> 32 cols
    const int lr   = lane & 15;
    const int hi   = lane >> 4;

    const long bz = blockIdx.z;
    const float* Ab = A + bz * bsA;
    const float* Bb = B + bz * bsB;
    float*       Cb = C + bz * bsC;

    const int m0 = blockIdx.x * GT_M;
    const int n0 = blockIdx.y * GT_N;

    v8f acc[2][2];
    const v8f vzero = {0.f,0.f,0.f,0.f,0.f,0.f,0.f,0.f};
    acc[0][0] = vzero; acc[0][1] = vzero; acc[1][0] = vzero; acc[1][1] = vzero;

    // each thread fills 16 A-halves and 16 B-halves per K-tile
    const int row = tid >> 1;           // 0..63
    const int kk0 = (tid & 1) * 16;     // 0 or 16

    // whole-tile-in-bounds => unchecked loads (true for all hot GEMMs here)
    const bool fast = (m0 + GT_M <= M) && (n0 + GT_N <= N) && (K % GT_K == 0);

    // per-thread base pointers (clamped so they are always dereferenceable)
    const int gmc = (m0 + row < M) ? (m0 + row) : (M - 1);
    const int gnc = (n0 + row < N) ? (n0 + row) : (N - 1);
    const bool rowAok = (m0 + row < M);
    const bool rowBok = (n0 + row < N);
    const float* pa = Ab + (long)gmc * rsA + (long)kk0 * csA;
    const float* pb = Bb + (long)kk0 * rsB + (long)gnc * csB;

    _Float16* wA = &sA[row * LDH + kk0];
    _Float16* wB = &sB[row * LDH + kk0];

    for (int k0 = 0; k0 < K; k0 += GT_K) {
        float ta[16], tb[16];
        if (fast) {
            if (k0 + GT_K < K) {   // prefetch next K-tile (global_prefetch_b8)
                __builtin_prefetch(pa + (long)GT_K * csA, 0, 1);
                __builtin_prefetch(pb + (long)GT_K * rsB, 0, 1);
            }
            #pragma unroll
            for (int j = 0; j < 16; ++j) ta[j] = pa[j * csA];
            #pragma unroll
            for (int j = 0; j < 16; ++j) tb[j] = pb[j * rsB];
        } else {
            #pragma unroll
            for (int j = 0; j < 16; ++j) {
                const int gk  = k0 + kk0 + j;
                const int gkc = (gk < K) ? gk : (K - 1);
                const int off = (gkc - k0 - kk0) * csA;   // 32-bit addressing
                ta[j] = pa[off];
                if (!(rowAok && gk < K)) ta[j] = 0.f;
            }
            #pragma unroll
            for (int j = 0; j < 16; ++j) {
                const int gk  = k0 + kk0 + j;
                const int gkc = (gk < K) ? gk : (K - 1);
                const int off = (gkc - k0 - kk0) * rsB;
                tb[j] = pb[off];
                if (!(rowBok && gk < K)) tb[j] = 0.f;
            }
        }
        pa += (long)GT_K * csA;
        pb += (long)GT_K * rsB;

        // packed f16 stores into LDS (compiler fuses into ds_store_b128)
        #pragma unroll
        for (int j = 0; j < 16; j += 2) {
            v2h p = {(_Float16)ta[j], (_Float16)ta[j + 1]};
            *(v2h*)&wA[j] = p;
        }
        #pragma unroll
        for (int j = 0; j < 16; j += 2) {
            v2h p = {(_Float16)tb[j], (_Float16)tb[j + 1]};
            *(v2h*)&wB[j] = p;
        }
        __syncthreads();

        // ---- fragments (ISA 7.12.2 layouts) ----
        Frag a[2], b[2];
        #pragma unroll
        for (int mt = 0; mt < 2; ++mt) {
            const _Float16* p = &sA[(wm * 32 + mt * 16 + lr) * LDH];
            // A 16x32 f16: lanes0-15 K=0..7 then 16..23; lanes16-31 K=8..15 then 24..31
            a[mt].h[0] = *(const v8h*)(p + hi * 8);
            a[mt].h[1] = *(const v8h*)(p + 16 + hi * 8);
        }
        #pragma unroll
        for (int nt = 0; nt < 2; ++nt) {
            const _Float16* p = &sB[(wn * 32 + nt * 16 + lr) * LDH];
            // B 32x16 f16: lanes0-15 K=0..7 then 16..23; lanes16-31 K=8..15 then 24..31
            b[nt].h[0] = *(const v8h*)(p + hi * 8);
            b[nt].h[1] = *(const v8h*)(p + 16 + hi * 8);
        }

        #pragma unroll
        for (int mt = 0; mt < 2; ++mt)
            #pragma unroll
            for (int nt = 0; nt < 2; ++nt)
                acc[mt][nt] = __builtin_amdgcn_wmma_f32_16x16x32_f16(
                    false, a[mt].f, false, b[nt].f,
                    (short)0, acc[mt][nt], false, false);
        __syncthreads();
    }

    // ---- epilogue: C 16x16 f32 layout: vgpr r -> row r (+8 for hi lanes) ----
    #pragma unroll
    for (int mt = 0; mt < 2; ++mt)
        #pragma unroll
        for (int nt = 0; nt < 2; ++nt)
            #pragma unroll
            for (int r = 0; r < 8; ++r) {
                const int rr = m0 + wm * 32 + mt * 16 + r + hi * 8;
                const int cc = n0 + wn * 32 + nt * 16 + lr;
                if (rr < M && cc < N)
                    Cb[(long)rr * rsC + (long)cc * csC] = acc[mt][nt][r];
            }
}

// ---------------------------------------------------------------------------
// Causal depthwise conv (K=4) along flattened sequence + SiLU
// xz: (B, L, 2*Di) row-major; u: (B, L, Di)
// ---------------------------------------------------------------------------
__global__ void conv_silu_kernel(const float* __restrict__ xz,
                                 const float* __restrict__ Wconv,
                                 const float* __restrict__ bconv,
                                 float* __restrict__ u,
                                 int B, int L, int Di, int ld_xz)
{
    const long idx = (long)blockIdx.x * blockDim.x + threadIdx.x;
    const long tot = (long)B * L * Di;
    if (idx >= tot) return;
    const int d = (int)(idx % Di);
    const long r = idx / Di;          // b*L + l
    const int l = (int)(r % L);
    const long bRow = r - l;          // b*L
    float acc = bconv[d];
    #pragma unroll
    for (int k = 0; k < 4; ++k) {
        const int ls = l + k - 3;
        if (ls >= 0) acc += Wconv[d * 4 + k] * xz[(bRow + ls) * (long)ld_xz + d];
    }
    u[idx] = acc / (1.f + __expf(-acc));   // silu
}

// ---------------------------------------------------------------------------
// Chunked parallel scan over the sequence. Recurrence h <- a*h + b with
//   a = exp(softplus(dt)*A[d,n]),  b = softplus(dt)*Bm[n]*u
// is associative: chunk aggregate (Aacc, Bacc) composes as
//   h_end = Aacc * h_start + Bacc.
// Pass 1: each (b,d,chunk) computes its aggregate (16x more parallelism).
// Pass 2: per (b,d), tiny sequential combine over NC chunks -> chunk h-init.
// Pass 3: each chunk re-runs locally from true h-init, produces gated output.
// ---------------------------------------------------------------------------
__global__ void scan_pass1_kernel(const float* __restrict__ dtraw,
                                  const float* __restrict__ bdt,
                                  const float* __restrict__ dbc,
                                  const float* __restrict__ u,
                                  const float* __restrict__ Alog,
                                  float* __restrict__ chA,
                                  float* __restrict__ chB,
                                  int B, int L, int Di, int R,
                                  int ld_dbc, int NC, int CS)
{
    const int idx = blockIdx.x * blockDim.x + threadIdx.x;
    if (idx >= B * Di * NC) return;
    const int c  = idx % NC;
    const int bd = idx / NC;          // b*Di + d
    const int d  = bd % Di;
    const int b  = bd / Di;

    float negA[16], Aacc[16], Bacc[16];
    #pragma unroll
    for (int n = 0; n < 16; ++n) {
        negA[n] = -__expf(Alog[d * 16 + n]);
        Aacc[n] = 1.f;
        Bacc[n] = 0.f;
    }
    const float bdt_d = bdt[d];
    const int l0 = c * CS;
    const int l1 = (l0 + CS < L) ? (l0 + CS) : L;

    for (int l = l0; l < l1; ++l) {
        const long r = (long)b * L + l;
        float dtv = dtraw[r * (long)Di + d] + bdt_d;
        dtv = (dtv > 20.f) ? dtv : __logf(1.f + __expf(dtv));
        const float dtu = dtv * u[r * (long)Di + d];
        const float* bc = dbc + r * (long)ld_dbc + R;   // Bm[0..15]
        #pragma unroll
        for (int n = 0; n < 16; ++n) {
            const float a = __expf(dtv * negA[n]);
            Bacc[n] = a * Bacc[n] + dtu * bc[n];
            Aacc[n] = a * Aacc[n];
        }
    }
    #pragma unroll
    for (int n = 0; n < 16; ++n) {
        chA[(long)idx * 16 + n] = Aacc[n];
        chB[(long)idx * 16 + n] = Bacc[n];
    }
}

__global__ void scan_pass2_kernel(const float* __restrict__ chA,
                                  const float* __restrict__ chB,
                                  float* __restrict__ hinit,
                                  int BD, int NC)
{
    const int bd = blockIdx.x * blockDim.x + threadIdx.x;
    if (bd >= BD) return;
    float h[16];
    #pragma unroll
    for (int n = 0; n < 16; ++n) h[n] = 0.f;
    for (int c = 0; c < NC; ++c) {
        const long base = ((long)bd * NC + c) * 16;
        #pragma unroll
        for (int n = 0; n < 16; ++n) {
            hinit[base + n] = h[n];
            h[n] = chA[base + n] * h[n] + chB[base + n];
        }
    }
}

__global__ void scan_pass3_kernel(const float* __restrict__ dtraw,
                                  const float* __restrict__ bdt,
                                  const float* __restrict__ dbc,
                                  const float* __restrict__ u,
                                  const float* __restrict__ xz,
                                  const float* __restrict__ Alog,
                                  const float* __restrict__ Dskip,
                                  const float* __restrict__ hinit,
                                  float* __restrict__ yg,
                                  int B, int L, int Di, int R,
                                  int ld_xz, int ld_dbc, int NC, int CS)
{
    const int idx = blockIdx.x * blockDim.x + threadIdx.x;
    if (idx >= B * Di * NC) return;
    const int c  = idx % NC;
    const int bd = idx / NC;
    const int d  = bd % Di;
    const int b  = bd / Di;

    float negA[16], h[16];
    #pragma unroll
    for (int n = 0; n < 16; ++n) {
        negA[n] = -__expf(Alog[d * 16 + n]);
        h[n] = hinit[(long)idx * 16 + n];
    }
    const float bdt_d = bdt[d];
    const float Dd    = Dskip[d];
    const int l0 = c * CS;
    const int l1 = (l0 + CS < L) ? (l0 + CS) : L;

    for (int l = l0; l < l1; ++l) {
        const long r = (long)b * L + l;
        float dtv = dtraw[r * (long)Di + d] + bdt_d;
        dtv = (dtv > 20.f) ? dtv : __logf(1.f + __expf(dtv));
        const float uv = u[r * (long)Di + d];
        const float dtu = dtv * uv;
        const float* bc = dbc + r * (long)ld_dbc + R;   // Bm[0..15], Cm[16..31]
        float accy = 0.f;
        #pragma unroll
        for (int n = 0; n < 16; ++n) {
            const float a = __expf(dtv * negA[n]);
            h[n] = a * h[n] + dtu * bc[n];
            accy += h[n] * bc[16 + n];
        }
        const float zv = xz[r * (long)ld_xz + Di + d];
        const float gate = zv / (1.f + __expf(-zv));
        yg[r * (long)Di + d] = (accy + uv * Dd) * gate;
    }
}

// ---------------------------------------------------------------------------
// out[b,d,l] = xin[b,d,l] + t[(b*L+l)*D + d]   (residual add + transpose)
// ---------------------------------------------------------------------------
__global__ void add_out_kernel(const float* __restrict__ xin,
                               const float* __restrict__ t,
                               float* __restrict__ out, int B, int D, int L)
{
    const long idx = (long)blockIdx.x * blockDim.x + threadIdx.x;
    const long tot = (long)B * D * L;
    if (idx >= tot) return;
    const int l = (int)(idx % L);
    const long r = idx / L;
    const int d = (int)(r % D);
    const int b = (int)(r / D);
    out[idx] = xin[idx] + t[((long)b * L + l) * (long)D + d];
}

// ---------------------------------------------------------------------------
// Bilinear resize (B,C,Hs,Ws) -> slot [cOff..cOff+C) of (B,Ctot,Ho,Wo)
// ---------------------------------------------------------------------------
__global__ void resize_kernel(const float* __restrict__ in, float* __restrict__ out,
                              int B, int C, int Hs, int Ws, int Ho, int Wo,
                              int Ctot, int cOff)
{
    const long idx = (long)blockIdx.x * blockDim.x + threadIdx.x;
    const long tot = (long)B * C * Ho * Wo;
    if (idx >= tot) return;
    const int x = (int)(idx % Wo);
    const int y = (int)((idx / Wo) % Ho);
    const int c = (int)((idx / ((long)Wo * Ho)) % C);
    const int b = (int)(idx / ((long)Wo * Ho * C));

    const float sy = (Ho > 1) ? (float)(Hs - 1) / (float)(Ho - 1) : 0.f;
    const float sx = (Wo > 1) ? (float)(Ws - 1) / (float)(Wo - 1) : 0.f;
    const float fy = y * sy, fx = x * sx;
    int y0 = (int)floorf(fy); int x0 = (int)floorf(fx);
    const float wy = fy - y0, wx = fx - x0;
    int y1 = min(y0 + 1, Hs - 1);
    int x1 = min(x0 + 1, Ws - 1);

    const float* p = in + ((long)b * C + c) * (long)Hs * Ws;
    const float v00 = p[y0 * Ws + x0], v01 = p[y0 * Ws + x1];
    const float v10 = p[y1 * Ws + x0], v11 = p[y1 * Ws + x1];
    const float top = v00 * (1.f - wx) + v01 * wx;
    const float bot = v10 * (1.f - wx) + v11 * wx;
    out[((long)b * Ctot + cOff + c) * (long)Ho * Wo + (long)y * Wo + x] =
        top * (1.f - wy) + bot * wy;
}

// ---------------------------------------------------------------------------
// out[b,o,l] = relu(((t[b][l*O+o] + pb[o]) - m[o]) * g[o]/sqrt(v[o]+eps) + bb[o])
// ---------------------------------------------------------------------------
__global__ void proj_bn_relu_kernel(const float* __restrict__ t,
                                    const float* __restrict__ pb,
                                    const float* __restrict__ g,
                                    const float* __restrict__ bb,
                                    const float* __restrict__ m,
                                    const float* __restrict__ v,
                                    float* __restrict__ out, int B, int L, int O)
{
    const long idx = (long)blockIdx.x * blockDim.x + threadIdx.x;
    const long tot = (long)B * O * L;
    if (idx >= tot) return;
    const int l = (int)(idx % L);
    const long r = idx / L;
    const int o = (int)(r % O);
    const int b = (int)(r / O);
    float val = t[((long)b * L + l) * (long)O + o] + pb[o];
    const float scale = g[o] * rsqrtf(v[o] + 1e-5f);
    val = (val - m[o]) * scale + bb[o];
    out[idx] = fmaxf(val, 0.f);
}

// ---------------------------------------------------------------------------
// Host-side helpers
// ---------------------------------------------------------------------------
static inline void launch_gemm(const float* A, const float* B, float* C,
                               int M, int N, int K,
                               int rsA, int csA, int rsB, int csB,
                               int rsC, int csC,
                               int batch, long bsA, long bsB, long bsC,
                               hipStream_t s)
{
    dim3 grid((M + GT_M - 1) / GT_M, (N + GT_N - 1) / GT_N, batch);
    gemm_wmma_kernel<<<grid, 128, 0, s>>>(A, B, C, M, N, K,
                                          rsA, csA, rsB, csB, rsC, csC,
                                          bsA, bsB, bsC);
}

struct MambaW {
    const float *Win, *Wconv, *bconv, *Wx, *Wdt, *bdt, *Alog, *Dskip, *Wout;
};

// out[b,d,l] = x[b,d,l] + mamba(x)[b,d,l]; x is (B, D, L) channel-major
static void run_mamba(const float* x, float* out, int B, int D, int L,
                      const MambaW& w,
                      float* XZ, float* U, float* DBC, float* DT, float* YG,
                      float* TOUT, float* CHA, float* CHB, float* HIN,
                      hipStream_t s)
{
    const int N  = 16;
    const int Di = 2 * D;
    const int R  = (D / 16 > 1) ? D / 16 : 1;
    const int ld_xz  = 2 * Di;
    const int ld_dbc = R + 2 * N;
    const int BL = B * L;

    // 1. xz = x^T @ Win   (batched: A[m=l,k=d] strided view of x)
    launch_gemm(x, w.Win, XZ, L, 2 * Di, D,
                /*A*/ 1, L, /*B*/ 2 * Di, 1,
                /*C*/ ld_xz, 1,
                B, (long)D * L, 0L, (long)L * ld_xz, s);

    // 2. u = silu(causal depthwise conv(xs))
    {
        long tot = (long)BL * Di;
        conv_silu_kernel<<<(int)((tot + 255) / 256), 256, 0, s>>>(
            XZ, w.Wconv, w.bconv, U, B, L, Di, ld_xz);
    }

    // 3. dbc = u @ Wx
    launch_gemm(U, w.Wx, DBC, BL, ld_dbc, Di,
                Di, 1, ld_dbc, 1, ld_dbc, 1,
                1, 0L, 0L, 0L, s);

    // 4. dt_raw = dbc[:, :R] @ Wdt
    launch_gemm(DBC, w.Wdt, DT, BL, Di, R,
                ld_dbc, 1, Di, 1, Di, 1,
                1, 0L, 0L, 0L, s);

    // 5. chunked parallel scan (softplus, D-skip, silu(z) gate fused)
    {
        const int NC = 16;
        const int CS = (L + NC - 1) / NC;
        const int tot1 = B * Di * NC;
        scan_pass1_kernel<<<(tot1 + 127) / 128, 128, 0, s>>>(
            DT, w.bdt, DBC, U, w.Alog, CHA, CHB, B, L, Di, R, ld_dbc, NC, CS);
        scan_pass2_kernel<<<(B * Di + 127) / 128, 128, 0, s>>>(
            CHA, CHB, HIN, B * Di, NC);
        scan_pass3_kernel<<<(tot1 + 127) / 128, 128, 0, s>>>(
            DT, w.bdt, DBC, U, XZ, w.Alog, w.Dskip, HIN, YG,
            B, L, Di, R, ld_xz, ld_dbc, NC, CS);
    }

    // 6. t = y @ Wout
    launch_gemm(YG, w.Wout, TOUT, BL, D, Di,
                Di, 1, D, 1, D, 1,
                1, 0L, 0L, 0L, s);

    // 7. out = x + t^T
    {
        long tot = (long)B * D * L;
        add_out_kernel<<<(int)((tot + 255) / 256), 256, 0, s>>>(x, TOUT, out, B, D, L);
    }
}

// ---------------------------------------------------------------------------
// Entry point
// ---------------------------------------------------------------------------
extern "C" void kernel_launch(void* const* d_in, const int* in_sizes, int n_in,
                              void* d_out, int out_size, void* d_ws, size_t ws_size,
                              hipStream_t stream)
{
    (void)in_sizes; (void)n_in; (void)out_size; (void)ws_size;
    const int B = 2, C = 64, Ho = 48, Wo = 48;
    const int Lc = Ho * Wo;              // 2304
    const int HS[4] = {48, 24, 12, 6};

    const float* feats[4] = {(const float*)d_in[0], (const float*)d_in[1],
                             (const float*)d_in[2], (const float*)d_in[3]};
    const float* lvl_Win   = (const float*)d_in[4];
    const float* lvl_Wconv = (const float*)d_in[5];
    const float* lvl_bconv = (const float*)d_in[6];
    const float* lvl_Wx    = (const float*)d_in[7];
    const float* lvl_Wdt   = (const float*)d_in[8];
    const float* lvl_bdt   = (const float*)d_in[9];
    const float* lvl_Alog  = (const float*)d_in[10];
    const float* lvl_D     = (const float*)d_in[11];
    const float* lvl_Wout  = (const float*)d_in[12];

    MambaW cr;
    cr.Win   = (const float*)d_in[13];
    cr.Wconv = (const float*)d_in[14];
    cr.bconv = (const float*)d_in[15];
    cr.Wx    = (const float*)d_in[16];
    cr.Wdt   = (const float*)d_in[17];
    cr.bdt   = (const float*)d_in[18];
    cr.Alog  = (const float*)d_in[19];
    cr.Dskip = (const float*)d_in[20];
    cr.Wout  = (const float*)d_in[21];

    const float* proj_W = (const float*)d_in[22];
    const float* proj_b = (const float*)d_in[23];
    const float* bn_g   = (const float*)d_in[24];
    const float* bn_b   = (const float*)d_in[25];
    const float* bn_m   = (const float*)d_in[26];
    const float* bn_v   = (const float*)d_in[27];

    // ---- workspace arena (floats), sized for the cross-level block ----
    float* ws  = (float*)d_ws;
    size_t off = 0;
    auto alloc = [&](size_t n) { float* p = ws + off; off += n; return p; };
    float* XZ   = alloc((size_t)B * Lc * 1024);   // xz (cross: 2*Di=1024)
    float* U    = alloc((size_t)B * Lc * 512);    // u
    float* DBC  = alloc((size_t)B * Lc * 48);     // dbc (cross ld=48)
    float* DT   = alloc((size_t)B * Lc * 512);    // dt_raw
    float* YG   = alloc((size_t)B * Lc * 512);    // gated y
    float* TOUT = alloc((size_t)B * Lc * 256);    // GEMM out / proj out
    float* FLVL = alloc((size_t)B * 64 * Lc);     // per-level residual result
    float* X    = alloc((size_t)B * 256 * Lc);    // concat buffer
    float* X2   = alloc((size_t)B * 256 * Lc);    // cross-level output
    float* CHA  = alloc((size_t)B * 512 * 16 * 16); // chunk aggregates (A)
    float* CHB  = alloc((size_t)B * 512 * 16 * 16); // chunk aggregates (B)
    float* HIN  = alloc((size_t)B * 512 * 16 * 16); // per-chunk h init

    // ---- per-level Mamba + resize into concat slots ----
    for (int i = 0; i < 4; ++i) {
        const int Hs = HS[i], Li = Hs * Hs;
        MambaW w;
        w.Win   = lvl_Win   + (size_t)i * 64 * 256;
        w.Wconv = lvl_Wconv + (size_t)i * 128 * 4;
        w.bconv = lvl_bconv + (size_t)i * 128;
        w.Wx    = lvl_Wx    + (size_t)i * 128 * 36;
        w.Wdt   = lvl_Wdt   + (size_t)i * 4 * 128;
        w.bdt   = lvl_bdt   + (size_t)i * 128;
        w.Alog  = lvl_Alog  + (size_t)i * 128 * 16;
        w.Dskip = lvl_D     + (size_t)i * 128;
        w.Wout  = lvl_Wout  + (size_t)i * 128 * 64;

        run_mamba(feats[i], FLVL, B, C, Li, w,
                  XZ, U, DBC, DT, YG, TOUT, CHA, CHB, HIN, stream);

        long tot = (long)B * C * Ho * Wo;
        resize_kernel<<<(int)((tot + 255) / 256), 256, 0, stream>>>(
            FLVL, X, B, C, Hs, Hs, Ho, Wo, 256, i * 64);
    }

    // ---- cross-level Mamba on concatenated features ----
    run_mamba(X, X2, B, 256, Lc, cr,
              XZ, U, DBC, DT, YG, TOUT, CHA, CHB, HIN, stream);

    // ---- projection (256 -> 64): y[b,o,l] = sum_c X2[b,c,l] * proj_W[o,c] ----
    launch_gemm(X2, proj_W, TOUT, Lc, 64, 256,
                /*A: [m=l,k=c] of (256,L)*/ 1, Lc,
                /*B: [k=c,n=o] of (64,256) row-major*/ 1, 256,
                /*C: (L,64)*/ 64, 1,
                B, (long)256 * Lc, 0L, (long)Lc * 64, stream);

    // ---- bias + batchnorm + relu into d_out (B,64,48,48) ----
    {
        long tot = (long)B * 64 * Lc;
        proj_bn_relu_kernel<<<(int)((tot + 255) / 256), 256, 0, stream>>>(
            TOUT, proj_b, bn_g, bn_b, bn_m, bn_v, (float*)d_out, B, Lc, 64);
    }
}
